// MFNO_79448305041934
// MI455X (gfx1250) — compile-verified, matrix operand order checked by reading, not run on hardware
//
#include <hip/hip_runtime.h>
#include <hip/hip_bf16.h>
#include <math.h>

// ---------------------------------------------------------------------------
// MFNO forward for gfx1250 (MI455X).  B=16, H=W=64, MODES=16, HID=12,
// LIFT=256, OUT=640, 4 spectral layers.
//  - Projection GEMM: v_wmma_f32_16x16x32_bf16, loop nest shaped to avoid
//    VGPR spills (ks8 outer / rt inner, 4 live accumulators, 1 live B-frag).
//  - t-tile staging in k_proj: Tensor Data Mover (tensor_load_to_lds) with a
//    hand-built D# descriptor, waited with s_wait_tensorcnt.
//  - FFTs are truncated-mode DFTs (only 16 kx x 32 ky modes are live).
// ---------------------------------------------------------------------------

typedef __bf16 bf16;
typedef __attribute__((ext_vector_type(16))) __bf16        v16bf;
typedef __attribute__((ext_vector_type(8)))  float         v8f;
typedef __attribute__((ext_vector_type(4)))  unsigned int  v4u;
typedef __attribute__((ext_vector_type(8)))  int           v8i_t;
typedef __attribute__((ext_vector_type(4)))  int           v4i_t;

#if __has_include(<hip/amd_detail/amd_gfx1250_TDM.h>)
#define TDM_6ARG 1
#else
#define TDM_6ARG 0
#endif

#define Bb   16
#define Hh   64
#define Ww   64
#define HID  12
#define LIFT 256
#define OUTC 640
#define MO   16
#define PIX  (Bb * Hh * Ww)          // 65536
#define IMG  (Hh * Ww)               // 4096

__device__ __forceinline__ float gelu_f(float x) {
    return 0.5f * x * (1.0f + erff(x * 0.70710678118654752f));
}

// ---------------------------------------------------------------------------
// 1) BatchNorm2d statistics over all of x (single channel).
// ---------------------------------------------------------------------------
__global__ void k_stats(const float* __restrict__ x, float* __restrict__ stats) {
    __shared__ float s1[1024];
    __shared__ float s2[1024];
    float a = 0.f, b = 0.f;
    for (int i = threadIdx.x; i < PIX; i += 1024) {
        float v = x[i];
        a += v; b += v * v;
    }
    s1[threadIdx.x] = a; s2[threadIdx.x] = b;
    __syncthreads();
    for (int off = 512; off > 0; off >>= 1) {
        if (threadIdx.x < off) {
            s1[threadIdx.x] += s1[threadIdx.x + off];
            s2[threadIdx.x] += s2[threadIdx.x + off];
        }
        __syncthreads();
    }
    if (threadIdx.x == 0) {
        float m = s1[0] * (1.0f / PIX);
        float v = s2[0] * (1.0f / PIX) - m * m;
        stats[0] = m;
        stats[1] = rsqrtf(v + 1e-5f);
    }
}

// ---------------------------------------------------------------------------
// 2) BN1d(condition) + branch MLP 2->50->50->1 (tanh) -> branch[16]
// ---------------------------------------------------------------------------
__global__ void k_branch(const float* __restrict__ cond,
                         const float* __restrict__ g1, const float* __restrict__ bb,
                         const float* __restrict__ w1, const float* __restrict__ b1,
                         const float* __restrict__ w2, const float* __restrict__ b2,
                         const float* __restrict__ w3, const float* __restrict__ b3,
                         float* __restrict__ branch) {
    __shared__ float cn[Bb * 2];
    __shared__ float h1[Bb * 50];
    __shared__ float h2[Bb * 50];
    __shared__ float cstat[4];
    int tid = threadIdx.x;
    if (tid < 2) {
        float s = 0.f, ss = 0.f;
        for (int i = 0; i < Bb; ++i) { float v = cond[i * 2 + tid]; s += v; ss += v * v; }
        float m = s * (1.0f / Bb);
        float var = ss * (1.0f / Bb) - m * m;
        cstat[tid] = m; cstat[2 + tid] = rsqrtf(var + 1e-5f);
    }
    __syncthreads();
    if (tid < Bb * 2) {
        int i = tid >> 1, k = tid & 1;
        cn[i * 2 + k] = (cond[i * 2 + k] - cstat[k]) * cstat[2 + k] * g1[k] + bb[k];
    }
    __syncthreads();
    for (int idx = tid; idx < Bb * 50; idx += blockDim.x) {
        int i = idx / 50, j = idx % 50;
        float s = b1[j];
        for (int k = 0; k < 2; ++k) s += cn[i * 2 + k] * w1[k * 50 + j];
        h1[i * 50 + j] = tanhf(s);
    }
    __syncthreads();
    for (int idx = tid; idx < Bb * 50; idx += blockDim.x) {
        int i = idx / 50, j = idx % 50;
        float s = b2[j];
        for (int k = 0; k < 50; ++k) s += h1[i * 50 + k] * w2[k * 50 + j];
        h2[i * 50 + j] = tanhf(s);
    }
    __syncthreads();
    if (tid < Bb) {
        float s = b3[0];
        for (int j = 0; j < 50; ++j) s += h2[tid * 50 + j] * w3[j];
        branch[tid] = tanhf(s);
    }
}

// ---------------------------------------------------------------------------
// 3) Lifting: xn -> gelu(xn*l1_w + l1_b)[256] -> @l2_w + l2_b -> t0[B,12,H,W]
// ---------------------------------------------------------------------------
__global__ void k_lift(const float* __restrict__ x, const float* __restrict__ stats,
                       const float* __restrict__ g, const float* __restrict__ be,
                       const float* __restrict__ l1w, const float* __restrict__ l1b,
                       const float* __restrict__ l2w, const float* __restrict__ l2b,
                       float* __restrict__ t0) {
    __shared__ float w2s[LIFT * HID];   // 12 KB
    __shared__ float w1s[LIFT];
    __shared__ float b1s[LIFT];
    int tid = threadIdx.x;
    for (int i = tid; i < LIFT * HID; i += 256) w2s[i] = l2w[i];
    for (int i = tid; i < LIFT; i += 256) { w1s[i] = l1w[i]; b1s[i] = l1b[i]; }
    __syncthreads();

    int pix = blockIdx.x * 256 + tid;
    float xn = (x[pix] - stats[0]) * stats[1] * g[0] + be[0];
    float acc[HID];
#pragma unroll
    for (int c = 0; c < HID; ++c) acc[c] = l2b[c];
    for (int j = 0; j < LIFT; ++j) {
        float hj = gelu_f(xn * w1s[j] + b1s[j]);
#pragma unroll
        for (int c = 0; c < HID; ++c) acc[c] += hj * w2s[j * HID + c];
    }
    int b = pix >> 12, rem = pix & (IMG - 1);
#pragma unroll
    for (int c = 0; c < HID; ++c) t0[((size_t)(b * HID + c) << 12) + rem] = acc[c];
}

// ---------------------------------------------------------------------------
// 4a) DFT along W: G[bc,h,kx] = sum_w t * e^{-2pi i kx w/64}, kx in [0,16)
// ---------------------------------------------------------------------------
__global__ void k_dftw(const float* __restrict__ t, float* __restrict__ G) {
    int gid = blockIdx.x * 256 + threadIdx.x;           // 192*64*16
    int kx = gid & 15;
    int h  = (gid >> 4) & 63;
    int bc = gid >> 10;
    const float* row = t + ((size_t)bc << 12) + (h << 6);
    float re = 0.f, im = 0.f;
    for (int w = 0; w < Ww; ++w) {
        float s, c;
        int p = (kx * w) & 63;
        __sincosf(-(6.283185307179586f / 64.f) * (float)p, &s, &c);
        float v = row[w];
        re += v * c; im += v * s;
    }
    size_t o = (((size_t)bc << 6) + h) * MO + kx;
    G[o * 2] = re; G[o * 2 + 1] = im;
}

// ---------------------------------------------------------------------------
// 4b) DFT along H on the 32 live rows: r<16 -> ky=r ; r>=16 -> ky=r+32
// ---------------------------------------------------------------------------
__global__ void k_dfth(const float* __restrict__ G, float* __restrict__ X) {
    int gid = blockIdx.x * 256 + threadIdx.x;           // 192*32*16
    int kx = gid & 15;
    int r  = (gid >> 4) & 31;
    int bc = gid >> 9;
    int ky = (r < 16) ? r : (r + 32);
    float re = 0.f, im = 0.f;
    for (int h = 0; h < Hh; ++h) {
        size_t o = (((size_t)bc << 6) + h) * MO + kx;
        float gr = G[o * 2], gi = G[o * 2 + 1];
        float s, c;
        int p = (ky * h) & 63;
        __sincosf(-(6.283185307179586f / 64.f) * (float)p, &s, &c);
        re += gr * c - gi * s;
        im += gr * s + gi * c;
    }
    size_t o = (((size_t)bc << 5) + r) * MO + kx;
    X[o * 2] = re; X[o * 2 + 1] = im;
}

// ---------------------------------------------------------------------------
// 4c) Spectral channel mix: O[b,o,r,kx] = sum_c X[b,c,r,kx] * W[c,o,r,kx]
// ---------------------------------------------------------------------------
__global__ void k_spec(const float* __restrict__ X,
                       const float* __restrict__ w1r, const float* __restrict__ w1i,
                       const float* __restrict__ w2r, const float* __restrict__ w2i,
                       float* __restrict__ O) {
    int gid = blockIdx.x * 256 + threadIdx.x;           // 16*12*32*16
    int kx = gid & 15;
    int r  = (gid >> 4) & 31;
    int bo = gid >> 9;
    int o  = bo % HID, b = bo / HID;
    const float* wr; const float* wi; int kw;
    if (r < 16) { wr = w1r; wi = w1i; kw = r; } else { wr = w2r; wi = w2i; kw = r - 16; }
    float re = 0.f, im = 0.f;
    for (int c = 0; c < HID; ++c) {
        size_t xo = ((((size_t)(b * HID + c)) << 5) + r) * MO + kx;
        float xr = X[xo * 2], xi = X[xo * 2 + 1];
        size_t wo = (((size_t)(c * HID + o)) * MO + kw) * MO + kx;
        float ar = wr[wo], ai = wi[wo];
        re += xr * ar - xi * ai;
        im += xr * ai + xi * ar;
    }
    size_t oo = (((size_t)bo << 5) + r) * MO + kx;
    O[oo * 2] = re; O[oo * 2 + 1] = im;
}

// ---------------------------------------------------------------------------
// 4d) Inverse DFT along H: S[bo,h,kx] = sum_r O * e^{+2pi i ky h/64}
// ---------------------------------------------------------------------------
__global__ void k_idfth(const float* __restrict__ O, float* __restrict__ S) {
    int gid = blockIdx.x * 256 + threadIdx.x;           // 192*64*16
    int kx = gid & 15;
    int h  = (gid >> 4) & 63;
    int bo = gid >> 10;
    float re = 0.f, im = 0.f;
    for (int r = 0; r < 32; ++r) {
        int ky = (r < 16) ? r : (r + 32);
        size_t oo = (((size_t)bo << 5) + r) * MO + kx;
        float orr = O[oo * 2], oi = O[oo * 2 + 1];
        float s, c;
        int p = (ky * h) & 63;
        __sincosf((6.283185307179586f / 64.f) * (float)p, &s, &c);
        re += orr * c - oi * s;
        im += orr * s + oi * c;
    }
    size_t o = (((size_t)bo << 6) + h) * MO + kx;
    S[o * 2] = re; S[o * 2 + 1] = im;
}

// ---------------------------------------------------------------------------
// 4e) Real reconstruction along W (kx>0 doubled, 1/4096 norm) + skip + GELU
// ---------------------------------------------------------------------------
__global__ void k_idftw(const float* __restrict__ S, const float* __restrict__ told,
                        const float* __restrict__ skw, const float* __restrict__ skb,
                        float* __restrict__ tnew, int do_gelu) {
    int gid = blockIdx.x * 256 + threadIdx.x;           // 192*64*64
    int w  = gid & 63;
    int h  = (gid >> 6) & 63;
    int bo = gid >> 12;
    int b = bo / HID, o = bo % HID;
    float acc = 0.f;
    for (int kx = 0; kx < MO; ++kx) {
        size_t so = (((size_t)bo << 6) + h) * MO + kx;
        float sr = S[so * 2], si = S[so * 2 + 1];
        float s, c;
        int p = (kx * w) & 63;
        __sincosf((6.283185307179586f / 64.f) * (float)p, &s, &c);
        float re = sr * c - si * s;
        acc += (kx == 0) ? re : 2.0f * re;
    }
    acc *= (1.0f / 4096.0f);
    float sk = skb[o];
    size_t base = ((size_t)b * HID << 12) + (h << 6) + w;
#pragma unroll
    for (int c = 0; c < HID; ++c) sk += told[base + ((size_t)c << 12)] * skw[c * HID + o];
    float v = acc + sk;
    if (do_gelu) v = gelu_f(v);
    tnew[(((size_t)bo) << 12) + (h << 6) + w] = v;
}

// ---------------------------------------------------------------------------
// 5) Projection: 12 -> 256 (GELU, LDS bf16) -> 640 via WMMA bf16, fused
//    p2_b, branch scale, transposed store [B,H,640,W].
//    t-tile is staged to LDS by the Tensor Data Mover.
// ---------------------------------------------------------------------------
union AFrag { v16bf v; uint4 q[2]; };

__global__ void k_proj(const float* __restrict__ t,
                       const float* __restrict__ p1w, const float* __restrict__ p1b,
                       const float* __restrict__ p2w, const float* __restrict__ p2b,
                       const float* __restrict__ branch, float* __restrict__ out) {
    __shared__ __align__(16) bf16 qlds[64 * LIFT];      // 32 KB
    __shared__ __align__(16) float tlds[HID * 64];      // 3 KB (TDM destination)
    __shared__ float p1s[HID * LIFT];                   // 12 KB
    __shared__ float p1bs[LIFT];

    int tid = threadIdx.x;
    int pix0 = blockIdx.x * 64;                         // one (b,h) row of 64 pixels
    int b = pix0 >> 12;
    int h = (pix0 >> 6) & 63;

    // --- TDM: DMA the 12x64 f32 tile (row stride 4096 elems) into tlds ----
    if ((tid >> 5) == 0) {                              // wave 0 only (uniform)
        size_t gaddr = (size_t)(const void*)(t + ((size_t)(b * HID) << 12) + (h << 6));
        unsigned laddr = (unsigned)(size_t)(void*)tlds; // LDS byte offset (addr[31:0])
        v4u g0;
        g0[0] = 1u;                                     // count=1, user descriptor
        g0[1] = laddr;                                  // lds_addr
        g0[2] = (unsigned)(gaddr & 0xFFFFFFFFu);        // global_addr[31:0]
        g0[3] = (unsigned)((gaddr >> 32) & 0x01FFFFFFu) // global_addr[56:32]
                | (2u << 30);                           // type=2 ("image")
        v8i_t g1;
        g1[0] = (int)(2u << 16);                        // data_size=4B
        g1[1] = (int)(64u << 16);                       // tensor_dim0=64 (lo16)
        g1[2] = (int)((unsigned)HID << 16);             // tensor_dim1=12 (lo16)
        g1[3] = (int)(64u << 16);                       // tile_dim0=64
        g1[4] = HID;                                    // tile_dim1=12
        g1[5] = 4096;                                   // tensor_dim0_stride lo32
        g1[6] = 0;
        g1[7] = 0;
        v4i_t gz = {0, 0, 0, 0};
#if TDM_6ARG
        v8i_t gz8 = {0, 0, 0, 0, 0, 0, 0, 0};
        __builtin_amdgcn_tensor_load_to_lds(g0, g1, gz, gz, gz8, 0);
#else
        __builtin_amdgcn_tensor_load_to_lds(g0, g1, gz, gz, 0);
#endif
        __builtin_amdgcn_s_wait_tensorcnt(0);
    }
    for (int i = tid; i < HID * LIFT; i += 256) p1s[i] = p1w[i];
    for (int i = tid; i < LIFT; i += 256) p1bs[i] = p1b[i];
    __syncthreads();

    // stage 1: q[p][ch] = gelu(p1_b + sum_c t*p1_w) in bf16
    for (int idx = tid; idx < 64 * LIFT; idx += 256) {
        int p = idx >> 8, ch = idx & 255;
        float s = p1bs[ch];
#pragma unroll
        for (int c = 0; c < HID; ++c) s += tlds[(c << 6) + p] * p1s[c * LIFT + ch];
        qlds[p * LIFT + ch] = (bf16)gelu_f(s);
    }
    __syncthreads();

    // stage 2: [64 x 256] @ [256 x 640] with v_wmma_f32_16x16x32_bf16
    // ks8 outer / rt inner: 4 live accumulators, ONE live B-fragment -> no spills.
    int wv   = tid >> 5;
    int lane = tid & 31;
    int rown = lane & 15;                 // A row within tile / B,D column
    int kb   = (lane >> 4) * 8;           // K sub-base per half-wave
    float brv = branch[b];

    for (int ct = wv; ct < OUTC / 16; ct += 8) {
        int ncol = ct * 16 + rown;
        __builtin_prefetch(p2w + ncol, 0, 1);           // global_prefetch_b8
        float bias = p2b[ncol];
        v8f acc0 = {}, acc1 = {}, acc2 = {}, acc3 = {};
#pragma unroll 1
        for (int ks8 = 0; ks8 < 8; ++ks8) {
            v16bf Bf;
#pragma unroll
            for (int i = 0; i < 16; ++i) {
                int k = ks8 * 32 + ((i < 8) ? (kb + i) : (16 + kb + (i - 8)));
                Bf[i] = (bf16)p2w[(size_t)k * OUTC + ncol];
            }
            const bf16* qk = qlds + ks8 * 32 + kb;
            AFrag a0, a1, a2, a3;
            a0.q[0] = *(const uint4*)(qk + (0 * 16 + rown) * LIFT);
            a0.q[1] = *(const uint4*)(qk + (0 * 16 + rown) * LIFT + 16);
            a1.q[0] = *(const uint4*)(qk + (1 * 16 + rown) * LIFT);
            a1.q[1] = *(const uint4*)(qk + (1 * 16 + rown) * LIFT + 16);
            a2.q[0] = *(const uint4*)(qk + (2 * 16 + rown) * LIFT);
            a2.q[1] = *(const uint4*)(qk + (2 * 16 + rown) * LIFT + 16);
            a3.q[0] = *(const uint4*)(qk + (3 * 16 + rown) * LIFT);
            a3.q[1] = *(const uint4*)(qk + (3 * 16 + rown) * LIFT + 16);
            acc0 = __builtin_amdgcn_wmma_f32_16x16x32_bf16(false, a0.v, false, Bf, (short)0, acc0, false, false);
            acc1 = __builtin_amdgcn_wmma_f32_16x16x32_bf16(false, a1.v, false, Bf, (short)0, acc1, false, false);
            acc2 = __builtin_amdgcn_wmma_f32_16x16x32_bf16(false, a2.v, false, Bf, (short)0, acc2, false, false);
            acc3 = __builtin_amdgcn_wmma_f32_16x16x32_bf16(false, a3.v, false, Bf, (short)0, acc3, false, false);
        }
        // writeback: lane has n=ncol, 8 consecutive w per rt tile
        float* dstb = out + (((size_t)(b * Hh + h) * OUTC + ncol) << 6) + (lane >> 4) * 8;
        v8f accs[4] = {acc0, acc1, acc2, acc3};
#pragma unroll
        for (int rt = 0; rt < 4; ++rt) {
            v8f a = accs[rt];
            float4 o0 = make_float4((a[0] + bias) * brv, (a[1] + bias) * brv,
                                    (a[2] + bias) * brv, (a[3] + bias) * brv);
            float4 o1 = make_float4((a[4] + bias) * brv, (a[5] + bias) * brv,
                                    (a[6] + bias) * brv, (a[7] + bias) * brv);
            float* dst = dstb + rt * 16;
            *(float4*)dst = o0;
            *(float4*)(dst + 4) = o1;
        }
    }
}

// ---------------------------------------------------------------------------
// Workspace layout (floats)
// ---------------------------------------------------------------------------
static const size_t OFF_STATS = 0;
static const size_t OFF_BR    = 32;
static const size_t OFF_T0    = 64;
static const size_t OFF_T1    = OFF_T0 + (size_t)Bb * HID * IMG;          // 786432
static const size_t OFF_G     = OFF_T1 + (size_t)Bb * HID * IMG;
static const size_t OFF_X     = OFF_G  + (size_t)Bb * HID * Hh * MO * 2;  // 393216
static const size_t OFF_O     = OFF_X  + (size_t)Bb * HID * 32 * MO * 2;  // 196608
static const size_t OFF_S     = OFF_O  + (size_t)Bb * HID * 32 * MO * 2;  // 196608

extern "C" void kernel_launch(void* const* d_in, const int* in_sizes, int n_in,
                              void* d_out, int out_size, void* d_ws, size_t ws_size,
                              hipStream_t stream) {
    const float* x    = (const float*)d_in[0];
    const float* cond = (const float*)d_in[1];
    const float* bn2g = (const float*)d_in[2];
    const float* bn2b = (const float*)d_in[3];
    const float* bn1g = (const float*)d_in[4];
    const float* bn1b = (const float*)d_in[5];
    const float* bw1  = (const float*)d_in[6];
    const float* bb1  = (const float*)d_in[7];
    const float* bw2  = (const float*)d_in[8];
    const float* bb2  = (const float*)d_in[9];
    const float* bw3  = (const float*)d_in[10];
    const float* bb3  = (const float*)d_in[11];
    const float* l1w  = (const float*)d_in[12];
    const float* l1b  = (const float*)d_in[13];
    const float* l2w  = (const float*)d_in[14];
    const float* l2b  = (const float*)d_in[15];
    const float* sw1r = (const float*)d_in[16];
    const float* sw1i = (const float*)d_in[17];
    const float* sw2r = (const float*)d_in[18];
    const float* sw2i = (const float*)d_in[19];
    const float* skw  = (const float*)d_in[20];
    const float* skb  = (const float*)d_in[21];
    const float* p1w  = (const float*)d_in[22];
    const float* p1b  = (const float*)d_in[23];
    const float* p2w  = (const float*)d_in[24];
    const float* p2b  = (const float*)d_in[25];

    float* out   = (float*)d_out;
    float* W     = (float*)d_ws;
    float* stats = W + OFF_STATS;
    float* br    = W + OFF_BR;
    float* t0    = W + OFF_T0;
    float* t1    = W + OFF_T1;
    float* G     = W + OFF_G;
    float* X     = W + OFF_X;
    float* O     = W + OFF_O;
    float* S     = W + OFF_S;

    k_stats<<<1, 1024, 0, stream>>>(x, stats);
    k_branch<<<1, 256, 0, stream>>>(cond, bn1g, bn1b, bw1, bb1, bw2, bb2, bw3, bb3, br);
    k_lift<<<PIX / 256, 256, 0, stream>>>(x, stats, bn2g, bn2b, l1w, l1b, l2w, l2b, t0);

    float* cur = t0; float* nxt = t1;
    for (int i = 0; i < 4; ++i) {
        const size_t swo = (size_t)i * HID * HID * MO * MO;   // 36864
        k_dftw <<<768,  256, 0, stream>>>(cur, G);
        k_dfth <<<384,  256, 0, stream>>>(G, X);
        k_spec <<<384,  256, 0, stream>>>(X, sw1r + swo, sw1i + swo, sw2r + swo, sw2i + swo, O);
        k_idfth<<<768,  256, 0, stream>>>(O, S);
        k_idftw<<<3072, 256, 0, stream>>>(S, cur, skw + (size_t)i * HID * HID,
                                          skb + (size_t)i * HID, nxt, (i < 3) ? 1 : 0);
        float* tmp = cur; cur = nxt; nxt = tmp;
    }

    k_proj<<<PIX / 64, 256, 0, stream>>>(cur, p1w, p1b, p2w, p2b, br, out);
}
